// ProjectedSS_Conv_68444598829446
// MI455X (gfx1250) — matrix operand order repaired and checked by reading, not compiled, not run on hardware
//
#include <hip/hip_runtime.h>
#include <stdint.h>

typedef __bf16 bf16_t;
typedef __attribute__((ext_vector_type(16))) __bf16 v16bf;
typedef __attribute__((ext_vector_type(8)))  float  v8f;

namespace {
constexpr int B   = 2;
constexpr int C   = 64;
constexpr int P   = 64;
constexpr int D   = 60;
constexpr int H   = 36;
constexpr int W   = 60;
constexpr int DHW = D * H * W;      // 129600
constexpr int NS  = B * DHW;        // 259200
constexpr int NTILE = NS / 16;      // 16200 spatial tiles of 16 columns
constexpr int MT_OFF = 6;           // 81 out channels -> 6 tiles of 16 (padded)
constexpr int MT_DEF = 4;           // 64 out channels -> 4 tiles
constexpr int ABLK_E = 27 * 2 * 512;   // bf16 elems of packed A per M-tile (27648)
constexpr int ABLK_Q = ABLK_E / 8;     // uint4 units (3456) = 55296 bytes
constexpr float EPSV = 1e-5f;
}

union Frag {
  v16bf  v;
  uint4  q[2];
  bf16_t h[16];
};

// ---------------------------------------------------------------------------
// x [B,C,D,H,W] f32  ->  xb [B,DHW,C] bf16  (channel innermost for b128 B-frags)
// ---------------------------------------------------------------------------
__global__ __launch_bounds__(256) void pack_x_kernel(const float* __restrict__ x,
                                                     bf16_t* __restrict__ xb) {
  int idx = blockIdx.x * 256 + threadIdx.x;          // over B*DHW*C
  if (idx >= B * DHW * C) return;
  int c = idx % C;
  int s = idx / C;                                   // b*DHW + sd
  int b = s / DHW, sd = s - b * DHW;
  xb[idx] = (bf16_t)x[(size_t)(b * C + c) * DHW + sd];
}

// ---------------------------------------------------------------------------
// Weight pre-shuffle into WMMA A-fragment order.
// A-layout (16-bit A 16x32): lane m = L%16, half = L>>4,
//   VGPR v (0..7): K = 16*half + 2v (+0/1), contiguous k (B frag uses same)
// Packed layout: [branch][mtile][tap][khalf][lane 0..31][16 bf16]
// (mtile outermost-but-one so one block's A is a single contiguous 54 KB blob)
// ---------------------------------------------------------------------------
__global__ __launch_bounds__(256) void pack_a_off_kernel(const float* __restrict__ off_w,
                                                         bf16_t* __restrict__ A) {
  int idx = blockIdx.x * 256 + threadIdx.x;          // 3*6*27*2*512
  if (idx >= 3 * MT_OFF * ABLK_E) return;
  int j    = idx & 15;
  int lane = (idx >> 4) & 31;
  int ch   = (idx >> 9) & 1;
  int t    = (idx >> 10) % 27;
  int mt   = (idx / (1024 * 27)) % MT_OFF;
  int i    = idx / (1024 * 27 * MT_OFF);
  int m = lane & 15, half = lane >> 4;
  int v = j >> 1, jj = j & 1;
  int k0 = 16 * half + 2 * v;
  int c  = ch * 32 + k0 + jj;
  int oc = mt * 16 + m;
  float val = 0.f;
  if (oc < 81) val = off_w[((size_t)((i * 81 + oc) * 64 + c)) * 27 + t];
  A[idx] = (bf16_t)val;
}

__global__ __launch_bounds__(256) void pack_a_def_kernel(const float* __restrict__ w,
                                                         bf16_t* __restrict__ A) {
  int idx = blockIdx.x * 256 + threadIdx.x;          // 3*4*27*2*512
  if (idx >= 3 * MT_DEF * ABLK_E) return;
  int j    = idx & 15;
  int lane = (idx >> 4) & 31;
  int ch   = (idx >> 9) & 1;
  int t    = (idx >> 10) % 27;
  int mt   = (idx / (1024 * 27)) % MT_DEF;
  int i    = idx / (1024 * 27 * MT_DEF);
  int m = lane & 15, half = lane >> 4;
  int v = j >> 1, jj = j & 1;
  int k0 = 16 * half + 2 * v;
  int c  = ch * 32 + k0 + jj;
  int oc = mt * 16 + m;
  A[idx] = (bf16_t)w[((size_t)((i * 64 + oc) * 64 + c)) * 27 + t];
}

// ---------------------------------------------------------------------------
// Offset conv (3x3x3, pad 1, 64 -> 81 ch) as implicit GEMM with bf16 WMMA.
// Block = 8 waves sharing one M-tile; packed A staged in LDS once per block.
// Two accumulators break the WMMA->WMMA RAW chain (avoids v_nop hazard slots).
// ---------------------------------------------------------------------------
__global__ __launch_bounds__(256) void offset_conv_kernel(const bf16_t* __restrict__ xb,
                                                          const bf16_t* __restrict__ Apk,
                                                          const float*  __restrict__ off_b,
                                                          float* __restrict__ offbuf,
                                                          int branch) {
  __shared__ uint4 shA[ABLK_Q];                      // 54 KB of LDS
  int mt = blockIdx.x % MT_OFF;                      // block-uniform M-tile
  int st = (blockIdx.x / MT_OFF) * 8 + (threadIdx.x >> 5);
  int lane = threadIdx.x & 31;

  // Stage this M-tile's packed weights: 3456 uint4 cooperative copy.
  {
    const uint4* gsrc =
        (const uint4*)(Apk + (size_t)(branch * MT_OFF + mt) * ABLK_E);
    for (int k = threadIdx.x; k < ABLK_Q; k += 256) shA[k] = gsrc[k];
  }
  __syncthreads();

  int n = lane & 15, half = lane >> 4;
  int scol = st * 16 + n;
  int bcol = scol / DHW;
  int sd   = scol - bcol * DHW;
  int d    = sd / (H * W);
  int rem  = sd - d * (H * W);
  int h    = rem / W;
  int ww0  = rem - h * W;

  v8f acc0 = {}, acc1 = {};
  int kd = -1, kh = -1, kw = -1;                     // incremental tap decode
  for (int t = 0; t < 27; ++t) {
    int zd = d + kd, zh = h + kh, zw = ww0 + kw;
    bool ok = (zd >= 0) & (zd < D) & (zh >= 0) & (zh < H) & (zw >= 0) & (zw < W);
    int row = ok ? (zd * H + zh) * W + zw : 0;
    const bf16_t* brow = xb + (size_t)(bcol * DHW + row) * C;
#pragma unroll
    for (int ch = 0; ch < 2; ++ch) {
      Frag fb;
      if (ok) {
        const uint4* src = (const uint4*)(brow + ch * 32 + half * 16);
        fb.q[0] = src[0];
        fb.q[1] = src[1];
      } else {
        fb.q[0] = make_uint4(0u, 0u, 0u, 0u);
        fb.q[1] = make_uint4(0u, 0u, 0u, 0u);
      }
      int abase = (t * 2 + ch) * 64 + lane * 2;      // uint4 units in LDS
      Frag fa;
      fa.q[0] = shA[abase];
      fa.q[1] = shA[abase + 1];
      if (ch == 0)
        acc0 = __builtin_amdgcn_wmma_f32_16x16x32_bf16(false, fa.v, false, fb.v,
                                                       (short)0, acc0, false, false);
      else
        acc1 = __builtin_amdgcn_wmma_f32_16x16x32_bf16(false, fa.v, false, fb.v,
                                                       (short)0, acc1, false, false);
    }
    if (++kw > 1) { kw = -1; if (++kh > 1) { kh = -1; ++kd; } }
  }
#pragma unroll
  for (int r = 0; r < 8; ++r) {
    int oc = mt * 16 + r + 8 * half;
    if (oc < 81)
      offbuf[(size_t)(bcol * 81 + oc) * DHW + sd] =
          acc0[r] + acc1[r] + off_b[branch * 81 + oc];
  }
}

// ---------------------------------------------------------------------------
// Deformable conv: per tap, trilinear-sample a 32-channel bf16 B-fragment in
// registers, multiply against LDS-staged packed weights via bf16 WMMA.
// ---------------------------------------------------------------------------
__global__ __launch_bounds__(256) void deform_conv_kernel(const bf16_t* __restrict__ xb,
                                                          const bf16_t* __restrict__ Apk,
                                                          const float*  __restrict__ offbuf,
                                                          float* __restrict__ z,
                                                          int branch, int dil) {
  __shared__ uint4 shA[ABLK_Q];                      // 54 KB of LDS
  int mt = blockIdx.x % MT_DEF;                      // block-uniform M-tile
  int st = (blockIdx.x / MT_DEF) * 8 + (threadIdx.x >> 5);
  int lane = threadIdx.x & 31;

  {
    const uint4* gsrc =
        (const uint4*)(Apk + (size_t)(branch * MT_DEF + mt) * ABLK_E);
    for (int k = threadIdx.x; k < ABLK_Q; k += 256) shA[k] = gsrc[k];
  }
  __syncthreads();

  int n = lane & 15, half = lane >> 4;
  int scol = st * 16 + n;
  int bcol = scol / DHW;
  int sd   = scol - bcol * DHW;
  int d    = sd / (H * W);
  int rem  = sd - d * (H * W);
  int h    = rem / W;
  int ww0  = rem - h * W;

  v8f acc0 = {}, acc1 = {};
  int kd = -1, kh = -1, kw = -1;                     // incremental tap decode
  size_t ob = (size_t)(bcol * 81) * DHW + sd;        // offset row for tap 0
  for (int t = 0; t < 27; ++t) {
    float od = offbuf[ob];
    float oh = offbuf[ob + DHW];
    float ow = offbuf[ob + 2 * (size_t)DHW];
    if (t < 26)                                       // warm next tap's offsets
      __builtin_prefetch(&offbuf[ob + 3 * (size_t)DHW], 0, 1);

    float pd = (float)d   + (float)(dil * kd) + od;
    float ph = (float)h   + (float)(dil * kh) + oh;
    float pw = (float)ww0 + (float)(dil * kw) + ow;
    float fd0 = floorf(pd), fh0 = floorf(ph), fw0 = floorf(pw);
    float fd = pd - fd0, fh = ph - fh0, fw = pw - fw0;
    int id0 = (int)fd0, ih0 = (int)fh0, iw0 = (int)fw0;

    float wts[8];
    int   rows[8];
#pragma unroll
    for (int dd = 0; dd < 2; ++dd) {
      int   idd = id0 + dd;
      float wd  = dd ? fd : 1.f - fd;
#pragma unroll
      for (int hh = 0; hh < 2; ++hh) {
        int   ihh = ih0 + hh;
        float wh  = hh ? fh : 1.f - fh;
#pragma unroll
        for (int wv = 0; wv < 2; ++wv) {
          int   iww = iw0 + wv;
          float wc  = wv ? fw : 1.f - fw;
          int q = dd * 4 + hh * 2 + wv;
          bool okc = (idd >= 0) & (idd < D) & (ihh >= 0) & (ihh < H) &
                     (iww >= 0) & (iww < W);
          rows[q] = okc ? (idd * H + ihh) * W + iww : -1;
          wts[q]  = wd * wh * wc;
        }
      }
    }

#pragma unroll
    for (int ch = 0; ch < 2; ++ch) {
      int coff = ch * 32 + half * 16;
      float facc[16];
#pragma unroll
      for (int j = 0; j < 16; ++j) facc[j] = 0.f;
#pragma unroll
      for (int q = 0; q < 8; ++q) {
        int row = rows[q];
        if (row >= 0) {
          const uint4* src =
              (const uint4*)(xb + (size_t)(bcol * DHW + row) * C + coff);
          Frag g;
          g.q[0] = src[0];
          g.q[1] = src[1];
          float wt = wts[q];
#pragma unroll
          for (int j = 0; j < 16; ++j) facc[j] = fmaf(wt, (float)g.h[j], facc[j]);
        }
      }
      Frag fb;
#pragma unroll
      for (int j = 0; j < 16; ++j) fb.h[j] = (bf16_t)facc[j];

      int abase = (t * 2 + ch) * 64 + lane * 2;      // uint4 units in LDS
      Frag fa;
      fa.q[0] = shA[abase];
      fa.q[1] = shA[abase + 1];
      if (ch == 0)
        acc0 = __builtin_amdgcn_wmma_f32_16x16x32_bf16(false, fa.v, false, fb.v,
                                                       (short)0, acc0, false, false);
      else
        acc1 = __builtin_amdgcn_wmma_f32_16x16x32_bf16(false, fa.v, false, fb.v,
                                                       (short)0, acc1, false, false);
    }
    ob += 3 * (size_t)DHW;
    if (++kw > 1) { kw = -1; if (++kh > 1) { kh = -1; ++kd; } }
  }
#pragma unroll
  for (int r = 0; r < 8; ++r) {
    int p = mt * 16 + r + 8 * half;
    z[(size_t)(bcol * P + p) * DHW + sd] = acc0[r] + acc1[r];
  }
}

// ---------------------------------------------------------------------------
// BatchNorm batch statistics: one block per channel (biased variance).
// ---------------------------------------------------------------------------
__global__ __launch_bounds__(256) void bn_stats_kernel(const float* __restrict__ z,
                                                       float* __restrict__ stats) {
  int p = blockIdx.x;
  float s = 0.f, ss = 0.f;
  for (int nn = threadIdx.x; nn < NS; nn += 256) {
    int b = nn / DHW, sd = nn - b * DHW;
    float v = z[(size_t)(b * P + p) * DHW + sd];
    s += v;
    ss += v * v;
  }
  __shared__ float sh[512];
  sh[threadIdx.x] = s;
  sh[256 + threadIdx.x] = ss;
  __syncthreads();
  for (int off = 128; off > 0; off >>= 1) {
    if (threadIdx.x < off) {
      sh[threadIdx.x] += sh[threadIdx.x + off];
      sh[256 + threadIdx.x] += sh[256 + threadIdx.x + off];
    }
    __syncthreads();
  }
  if (threadIdx.x == 0) {
    float mean = sh[0] / (float)NS;
    float var  = sh[256] / (float)NS - mean * mean;
    stats[p]      = mean;
    stats[64 + p] = var;
  }
}

// ---------------------------------------------------------------------------
// BN + ReLU, accumulate branches into d_out (branch 0 writes, others add).
// ---------------------------------------------------------------------------
__global__ __launch_bounds__(256) void bn_relu_add_kernel(const float* __restrict__ z,
                                                          const float* __restrict__ stats,
                                                          const float* __restrict__ gamma,
                                                          const float* __restrict__ beta,
                                                          float* __restrict__ out,
                                                          int branch) {
  int idx = blockIdx.x * 256 + threadIdx.x;
  if (idx >= B * P * DHW) return;
  int p = (idx / DHW) % P;
  float mean = stats[p], var = stats[64 + p];
  float yn = (z[idx] - mean) * rsqrtf(var + EPSV);
  float v = gamma[branch * P + p] * yn + beta[branch * P + p];
  v = fmaxf(v, 0.f);
  if (branch == 0) out[idx] = v;
  else             out[idx] += v;
}

// ---------------------------------------------------------------------------
extern "C" void kernel_launch(void* const* d_in, const int* in_sizes, int n_in,
                              void* d_out, int out_size, void* d_ws, size_t ws_size,
                              hipStream_t stream) {
  const float* x      = (const float*)d_in[0];  // [2,64,60,36,60]
  const float* off_w  = (const float*)d_in[1];  // [3,81,64,3,3,3]
  const float* off_b  = (const float*)d_in[2];  // [3,81]
  const float* w      = (const float*)d_in[3];  // [3,64,64,3,3,3]
  const float* gamma  = (const float*)d_in[4];  // [3,64]
  const float* beta   = (const float*)d_in[5];  // [3,64]
  float* out = (float*)d_out;

  // Carve workspace (all regions fully rewritten every call -> deterministic).
  auto up = [](size_t v) { return (v + 255) & ~(size_t)255; };
  char* ws = (char*)d_ws;
  size_t o = 0;
  bf16_t* xb   = (bf16_t*)(ws + o); o = up(o + (size_t)B * DHW * C * sizeof(bf16_t));
  bf16_t* Aoff = (bf16_t*)(ws + o); o = up(o + (size_t)3 * MT_OFF * ABLK_E * sizeof(bf16_t));
  bf16_t* Adef = (bf16_t*)(ws + o); o = up(o + (size_t)3 * MT_DEF * ABLK_E * sizeof(bf16_t));
  float* offbuf = (float*)(ws + o); o = up(o + (size_t)B * 81 * DHW * sizeof(float));
  float* zbuf   = (float*)(ws + o); o = up(o + (size_t)B * P * DHW * sizeof(float));
  float* stats  = (float*)(ws + o); o = up(o + 128 * sizeof(float));
  (void)ws_size; (void)in_sizes; (void)n_in; (void)out_size;

  // One-time (per call) repacks
  {
    int tot = B * DHW * C;
    pack_x_kernel<<<(tot + 255) / 256, 256, 0, stream>>>(x, xb);
  }
  {
    int tot = 3 * MT_OFF * ABLK_E;
    pack_a_off_kernel<<<(tot + 255) / 256, 256, 0, stream>>>(off_w, Aoff);
  }
  {
    int tot = 3 * MT_DEF * ABLK_E;
    pack_a_def_kernel<<<(tot + 255) / 256, 256, 0, stream>>>(w, Adef);
  }

  const int dils[3] = {1, 2, 3};
  for (int i = 0; i < 3; ++i) {
    {
      int blocks = (NTILE / 8) * MT_OFF;            // 2025 * 6 = 12150
      offset_conv_kernel<<<blocks, 256, 0, stream>>>(xb, Aoff, off_b, offbuf, i);
    }
    {
      int blocks = (NTILE / 8) * MT_DEF;            // 2025 * 4 = 8100
      deform_conv_kernel<<<blocks, 256, 0, stream>>>(xb, Adef, offbuf, zbuf, i, dils[i]);
    }
    bn_stats_kernel<<<P, 256, 0, stream>>>(zbuf, stats);
    {
      int tot = B * P * DHW;
      bn_relu_add_kernel<<<(tot + 255) / 256, 256, 0, stream>>>(zbuf, stats, gamma, beta, out, i);
    }
  }
}
